// MEO_48584670053013
// MI455X (gfx1250) — compile-verified, matrix operand order checked by reading, not run on hardware
//
#include <hip/hip_runtime.h>
#include <hip/hip_bf16.h>

typedef __attribute__((ext_vector_type(2)))  float  v2f;
typedef __attribute__((ext_vector_type(4)))  float  v4f;
typedef __attribute__((ext_vector_type(8)))  float  v8f;
typedef __attribute__((ext_vector_type(4)))  __bf16 v4bf;
typedef __attribute__((ext_vector_type(8)))  __bf16 v8bf;
typedef __attribute__((ext_vector_type(16))) __bf16 v16bf;

__device__ __forceinline__ v8f wmma_f32(v2f a, v2f b, v8f c) {
  return __builtin_amdgcn_wmma_f32_16x16x4_f32(false, a, false, b, (short)0, c,
                                               false, false);
}
__device__ __forceinline__ v8f wmma_bf16(v16bf a, v16bf b, v8f c) {
  return __builtin_amdgcn_wmma_f32_16x16x32_bf16(false, a, false, b, (short)0,
                                                 c, false, false);
}
__device__ __forceinline__ v16bf cat16(v8bf a, v8bf b) {
  return __builtin_shufflevector(a, b, 0, 1, 2, 3, 4, 5, 6, 7, 8, 9, 10, 11,
                                 12, 13, 14, 15);
}

// -------------------------------------------------------------------------
// Kernel A: x' = x + relu(x@Wd^T + bd)@Wu^T + bu  (exact f32 WMMA path)
// Writes x' as bf16 hi/lo split + per-16-row column sums for pooling.
// block = 128 threads (4 waves), wave owns 16 rows; grid = 16384/64
// -------------------------------------------------------------------------
__global__ __launch_bounds__(128) void adapter_kernel(
    const float* __restrict__ x, const float* __restrict__ Wd,
    const float* __restrict__ bd, const float* __restrict__ Wu,
    const float* __restrict__ bu, __bf16* __restrict__ xhi,
    __bf16* __restrict__ xlo, float* __restrict__ partial) {
  const int tid = threadIdx.x;
  const int wave = tid >> 5;
  const int lane = tid & 31;
  const int lm = lane & 15;   // M (or N) index within fragment
  const int lh = lane >> 4;   // K half selector
  const int row0 = blockIdx.x * 64 + wave * 16;  // global flat row (b*S + s)

  __shared__ float hbuf[4 * 16 * 18];
  float* hb = &hbuf[wave * 16 * 18];

  // ---- h = x_tile(16x1024) @ Wd^T(1024x16), K=1024 ----
  v8f hc = {};
  const float* xrow = x + (size_t)(row0 + lm) * 1024 + 2 * lh;
  const float* wdrow = Wd + (size_t)lm * 1024 + 2 * lh;
#pragma unroll 4
  for (int k = 0; k < 1024; k += 4) {
    v2f a = *(const v2f*)(xrow + k);
    v2f b = *(const v2f*)(wdrow + k);
    hc = wmma_f32(a, b, hc);
  }
  const float bdv = bd[lm];
#pragma unroll
  for (int v = 0; v < 8; ++v) {
    float t = hc[v] + bdv;
    hc[v] = t > 0.0f ? t : 0.0f;
  }
  // scatter C-layout -> row-major 16x16 in LDS (stride 18: conflict-free)
#pragma unroll
  for (int v = 0; v < 8; ++v) hb[(v + 8 * lh) * 18 + lm] = hc[v];
  __syncthreads();
  v2f af[4];
#pragma unroll
  for (int kk = 0; kk < 4; ++kk)
    af[kk] = *(const v2f*)(hb + lm * 18 + kk * 4 + 2 * lh);

  // ---- u = h'(16x16) @ Wu^T(16x1024), fused epilogue + bf16 split ----
  const int rtile = row0 >> 4;  // 0..1023
  for (int nt = 0; nt < 64; ++nt) {
    const int n0 = nt * 16;
    v8f c = {};
#pragma unroll
    for (int kk = 0; kk < 4; ++kk) {
      v2f b = *(const v2f*)(Wu + (size_t)(n0 + lm) * 16 + kk * 4 + 2 * lh);
      c = wmma_f32(af[kk], b, c);
    }
    const float buv = bu[n0 + lm];
    float csum = 0.0f;
#pragma unroll
    for (int v = 0; v < 8; ++v) {
      const int row = row0 + v + 8 * lh;
      const size_t idx = (size_t)row * 1024 + n0 + lm;
      const float val = c[v] + x[idx] + buv;
      const __bf16 hi = (__bf16)val;            // RNE
      xhi[idx] = hi;
      xlo[idx] = (__bf16)(val - (float)hi);
      csum += val;
    }
    csum += __shfl_xor(csum, 16, 32);  // combine the two 8-row halves
    if (lh == 0) partial[(size_t)rtile * 1024 + n0 + lm] = csum;
  }
}

// -------------------------------------------------------------------------
// Kernel B1: pooled[b][d] = mean_s x'[b,s,d]  (fixed-order, deterministic)
// -------------------------------------------------------------------------
__global__ __launch_bounds__(256) void pool_reduce(
    const float* __restrict__ partial, float* __restrict__ pooled) {
  const int idx = blockIdx.x * 256 + threadIdx.x;  // 0..8191
  const int b = idx >> 10, d = idx & 1023;
  const float* p = partial + (size_t)b * 128 * 1024 + d;
  float s = 0.0f;
  for (int t = 0; t < 128; ++t) s += p[(size_t)t * 1024];
  pooled[idx] = s * (1.0f / 2048.0f);
}

// -------------------------------------------------------------------------
// Kernel B2: logits, top-2 softmax gates, CV-squared aux loss
// -------------------------------------------------------------------------
__global__ __launch_bounds__(64) void gate_kernel(
    const float* __restrict__ pooled, const float* __restrict__ w_gate,
    float* __restrict__ gates_out, int* __restrict__ eidx_out,
    float* __restrict__ loss_out) {
  __shared__ float logits[64];
  __shared__ float gbuf[16];
  __shared__ int ebuf[16];
  const int t = threadIdx.x;
  {  // logits[b][e] = pooled[b] . w_gate[:,e]
    const int b = t >> 3, e = t & 7;
    float s = 0.0f;
    for (int d = 0; d < 1024; ++d) s += pooled[b * 1024 + d] * w_gate[d * 8 + e];
    logits[t] = s;
  }
  __syncthreads();
  if (t < 8) {
    const int b = t;
    const float* lg = &logits[b * 8];
    int i0 = 0;
    float m0 = lg[0];
    for (int e = 1; e < 8; ++e)
      if (lg[e] > m0) { m0 = lg[e]; i0 = e; }   // first max wins (JAX tie-break)
    int i1 = 0;
    float m1 = -1e30f;
    for (int e = 0; e < 8; ++e)
      if (e != i0 && lg[e] > m1) { m1 = lg[e]; i1 = e; }
    const float e1 = expf(m1 - m0);             // e0 = exp(0) = 1
    const float z = 1.0f + e1;
    const float g0 = 1.0f / z, g1 = e1 / z;
    gbuf[b * 2 + 0] = g0; gbuf[b * 2 + 1] = g1;
    ebuf[b * 2 + 0] = i0; ebuf[b * 2 + 1] = i1;
    gates_out[b * 2 + 0] = g0; gates_out[b * 2 + 1] = g1;
    eidx_out[b * 2 + 0] = i0;  eidx_out[b * 2 + 1] = i1;
  }
  __syncthreads();
  if (t == 0) {
    float imp[8] = {}, load[8] = {};
    for (int b = 0; b < 8; ++b) {
      imp[ebuf[b * 2 + 0]] += gbuf[b * 2 + 0]; load[ebuf[b * 2 + 0]] += 1.0f;
      imp[ebuf[b * 2 + 1]] += gbuf[b * 2 + 1]; load[ebuf[b * 2 + 1]] += 1.0f;
    }
    float loss = 0.0f;
    for (int which = 0; which < 2; ++which) {
      const float* v = which ? load : imp;
      float mean = 0.0f;
      for (int e = 0; e < 8; ++e) mean += v[e];
      mean *= (1.0f / 8.0f);
      float var = 0.0f;
      for (int e = 0; e < 8; ++e) { float d = v[e] - mean; var += d * d; }
      var *= (1.0f / 7.0f);  // ddof=1
      loss += var / (mean * mean + 1e-10f);
    }
    *loss_out = loss * 0.01f;
  }
}

// -------------------------------------------------------------------------
// Kernel B3: expert_w[b] = g0*W[e0] + g1*W[e1], split to bf16 hi/lo (once)
// -------------------------------------------------------------------------
__global__ __launch_bounds__(256) void combine_split(
    const float* __restrict__ weight, const float* __restrict__ gates,
    const int* __restrict__ eidx, __bf16* __restrict__ wch,
    __bf16* __restrict__ wcl) {
  const size_t i = ((size_t)blockIdx.x * 256 + threadIdx.x) * 4;  // 8*2^20 elems
  const int b = (int)(i >> 20);
  const size_t r = i & 1048575;
  const float g0 = gates[b * 2 + 0], g1 = gates[b * 2 + 1];
  const float* w0 = weight + (size_t)eidx[b * 2 + 0] * 1048576 + r;
  const float* w1 = weight + (size_t)eidx[b * 2 + 1] * 1048576 + r;
  v4f v = g0 * (*(const v4f*)w0) + g1 * (*(const v4f*)w1);
  v4bf hi, lo;
#pragma unroll
  for (int j = 0; j < 4; ++j) {
    hi[j] = (__bf16)v[j];
    lo[j] = (__bf16)(v[j] - (float)hi[j]);
  }
  *(v4bf*)(wch + i) = hi;
  *(v4bf*)(wcl + i) = lo;
}

// -------------------------------------------------------------------------
// Kernel C: y[b] = x'[b](2048x1024) @ expert_w[b]^T via bf16x3 split WMMA
// block = 256 threads (8 waves, 4x2) -> 256x128 output tile; wave -> 64x64
// LDS rows padded to 40 bf16 (80 B: 16B-aligned; 20r mod 64 = multiples of 4
// -> b128 lane spans tile all 64 banks, conflict-free). LDS total = 60 KB.
// -------------------------------------------------------------------------
#define KSTR 40
__global__ __launch_bounds__(256) void moe_gemm(
    const __bf16* __restrict__ xhi, const __bf16* __restrict__ xlo,
    const __bf16* __restrict__ wch, const __bf16* __restrict__ wcl,
    float* __restrict__ y) {
  const int b = blockIdx.z;
  const int s0 = blockIdx.y * 256;
  const int o0 = blockIdx.x * 128;
  const int tid = threadIdx.x;
  const int wave = tid >> 5, lane = tid & 31, lm = lane & 15, lh = lane >> 4;
  const int wm = wave >> 1;  // 0..3: 64-row band
  const int wn = wave & 1;   // 0..1: 64-col band

  __shared__ __bf16 xsh[256 * KSTR], xsl[256 * KSTR];  // x' tile 256x32 hi/lo
  __shared__ __bf16 wsh[128 * KSTR], wsl[128 * KSTR];  // W  tile 128x32 hi/lo

  const size_t xoff = (size_t)b * 2048 * 1024;
  const size_t woff = (size_t)b * 1024 * 1024;

  v8f c[4][4] = {};  // 64 rows x 64 cols per wave

  for (int kb = 0; kb < 1024; kb += 32) {
    __syncthreads();
    // stage x' tile: 256 rows x 32 cols, hi+lo (4 v8bf each per thread)
#pragma unroll
    for (int i = 0; i < 4; ++i) {
      const int idx = tid + i * 256;  // 0..1023
      const int row = idx >> 2, c8 = (idx & 3) << 3;
      const size_t g = xoff + (size_t)(s0 + row) * 1024 + kb + c8;
      *(v8bf*)(&xsh[row * KSTR + c8]) = *(const v8bf*)(xhi + g);
      *(v8bf*)(&xsl[row * KSTR + c8]) = *(const v8bf*)(xlo + g);
    }
    // stage weight tile: 128 rows x 32 cols, hi+lo (2 v8bf each per thread)
#pragma unroll
    for (int i = 0; i < 2; ++i) {
      const int idx = tid + i * 256;  // 0..511
      const int row = idx >> 2, c8 = (idx & 3) << 3;
      const size_t g = woff + (size_t)(o0 + row) * 1024 + kb + c8;
      *(v8bf*)(&wsh[row * KSTR + c8]) = *(const v8bf*)(wch + g);
      *(v8bf*)(&wsl[row * KSTR + c8]) = *(const v8bf*)(wcl + g);
    }
    __syncthreads();

    // A fragments: lane holds K in [8*lh, +8) and [16+8*lh, +8) (ISA 16-bit
    // A layout) -> two contiguous 16B runs from the row-major LDS row.
    v16bf ah[4], al[4];
#pragma unroll
    for (int m = 0; m < 4; ++m) {
      const __bf16* ph = &xsh[(wm * 64 + m * 16 + lm) * KSTR + 8 * lh];
      const __bf16* pl = &xsl[(wm * 64 + m * 16 + lm) * KSTR + 8 * lh];
      ah[m] = cat16(*(const v8bf*)ph, *(const v8bf*)(ph + 16));
      al[m] = cat16(*(const v8bf*)pl, *(const v8bf*)(pl + 16));
    }
#pragma unroll
    for (int n = 0; n < 4; ++n) {
      const __bf16* ph = &wsh[(wn * 64 + n * 16 + lm) * KSTR + 8 * lh];
      const __bf16* pl = &wsl[(wn * 64 + n * 16 + lm) * KSTR + 8 * lh];
      v16bf bh = cat16(*(const v8bf*)ph, *(const v8bf*)(ph + 16));
      v16bf bl = cat16(*(const v8bf*)pl, *(const v8bf*)(pl + 16));
#pragma unroll
      for (int m = 0; m < 4; ++m) {
        c[m][n] = wmma_bf16(ah[m], bh, c[m][n]);  // hi*hi
        c[m][n] = wmma_bf16(ah[m], bl, c[m][n]);  // hi*lo
        c[m][n] = wmma_bf16(al[m], bh, c[m][n]);  // lo*hi
      }
    }
  }
  // epilogue
#pragma unroll
  for (int m = 0; m < 4; ++m) {
#pragma unroll
    for (int n = 0; n < 4; ++n) {
#pragma unroll
      for (int v = 0; v < 8; ++v) {
        const int row = s0 + wm * 64 + m * 16 + v + 8 * lh;
        const int col = o0 + wn * 64 + n * 16 + lm;
        y[((size_t)b * 2048 + row) * 1024 + col] = c[m][n][v];
      }
    }
  }
}

// -------------------------------------------------------------------------
extern "C" void kernel_launch(void* const* d_in, const int* in_sizes, int n_in,
                              void* d_out, int out_size, void* d_ws,
                              size_t ws_size, hipStream_t stream) {
  const float* x = (const float*)d_in[0];       // (8,2048,1024)
  const float* w_gate = (const float*)d_in[1];  // (1024,8)
  const float* Wd = (const float*)d_in[2];      // (16,1024)
  const float* bd = (const float*)d_in[3];      // (16,)
  const float* Wu = (const float*)d_in[4];      // (1024,16)
  const float* bu = (const float*)d_in[5];      // (1024,)
  const float* weight = (const float*)d_in[6];  // (8,1024,1024)

  float* y = (float*)d_out;            // 16777216 floats
  float* loss = y + 16777216;          // +1 float

  char* ws = (char*)d_ws;
  __bf16* xhi = (__bf16*)(ws + 0);                  //  32 MB
  __bf16* xlo = (__bf16*)(ws + (size_t)33554432);   //  32 MB
  __bf16* wch = (__bf16*)(ws + (size_t)67108864);   //  16 MB
  __bf16* wcl = (__bf16*)(ws + (size_t)83886080);   //  16 MB
  float* partial = (float*)(ws + (size_t)100663296);//   4 MB
  float* pooled = (float*)(ws + (size_t)104857600); //  32 KB
  float* gates = (float*)(ws + (size_t)104890368);  //  64 B
  int* eidx = (int*)(ws + (size_t)104890432);       //  64 B

  adapter_kernel<<<256, 128, 0, stream>>>(x, Wd, bd, Wu, bu, xhi, xlo, partial);
  pool_reduce<<<32, 256, 0, stream>>>(partial, pooled);
  gate_kernel<<<1, 64, 0, stream>>>(pooled, w_gate, gates, eidx, loss);
  combine_split<<<8192, 256, 0, stream>>>(weight, gates, eidx, wch, wcl);
  dim3 grid(8, 8, 8);  // (o_tiles, s_tiles, b)
  moe_gemm<<<grid, 256, 0, stream>>>(xhi, xlo, wch, wcl, y);
}
